// SPKT_50379966382743
// MI455X (gfx1250) — compile-verified, matrix operand order checked by reading, not run on hardware
//
#include <hip/hip_runtime.h>
#include <hip/hip_bf16.h>
#include <math.h>

#define B_ 8
#define S_ 1024
#define D_ 1024
#define H_ 8
#define DK_ 128
#define FF_ 4096
#define L_ 2
#define M_ (B_ * S_)          // 8192 rows in every GEMM
#define NEGINF (-1e32f)

typedef __bf16 bf16_t;
typedef __attribute__((ext_vector_type(16))) __bf16 v16bf;
typedef __attribute__((ext_vector_type(8)))  float  v8f;

static_assert(sizeof(v16bf) == 32, "v16bf must be 32B");

// CDNA5 async global->LDS DMA (ASYNCcnt) if the toolchain exposes it.
// Probe-confirmed prototype: (int4 as1* gsrc, int4 as3* ldst, imm off, imm cpol)
#if __has_builtin(__builtin_amdgcn_global_load_async_to_lds_b128)
#define USE_ASYNC_LDS 1
typedef int v4i_gcc __attribute__((__vector_size__(16)));
typedef __attribute__((address_space(1))) v4i_gcc as1_v4i;
typedef __attribute__((address_space(3))) v4i_gcc as3_v4i;
#else
#define USE_ASYNC_LDS 0
#endif

__device__ __forceinline__ void copy16_to_lds(const void* g, void* l) {
#if USE_ASYNC_LDS
  // Generic LDS addresses keep the LDS byte offset in the low 32 bits
  // (ISA 10.2 aperture rules), so truncation yields a valid as3 pointer.
  __builtin_amdgcn_global_load_async_to_lds_b128(
      (as1_v4i*)(unsigned long long)g,
      (as3_v4i*)(unsigned int)(unsigned long long)l, 0, 0);
#else
  *(uint4*)l = *(const uint4*)g;
#endif
}

__device__ __forceinline__ void wait_async_lds() {
#if USE_ASYNC_LDS
#if __has_builtin(__builtin_amdgcn_s_wait_asynccnt)
  __builtin_amdgcn_s_wait_asynccnt(0);
#else
  asm volatile("s_wait_asynccnt 0" ::: "memory");
#endif
#endif
}

__device__ __forceinline__ bf16_t f2bf(float f) {
  unsigned u = __builtin_bit_cast(unsigned, f);
  unsigned r = u + 0x7fffu + ((u >> 16) & 1u);   // round-to-nearest-even
  unsigned short h = (unsigned short)(r >> 16);
  return __builtin_bit_cast(bf16_t, h);
}

struct FragU { union { v16bf v; uint4 u[2]; }; };

__device__ __forceinline__ v16bf frag_load(const bf16_t* p) {
  FragU f;
  f.u[0] = *(const uint4*)(p);
  f.u[1] = *(const uint4*)(p + 8);
  return f.v;
}

__device__ __forceinline__ v8f wmma_bf16(v16bf a, v16bf b, v8f c) {
  // (neg_a, A, neg_b, B, c_mod, C, reuse_a, reuse_b)
  return __builtin_amdgcn_wmma_f32_16x16x32_bf16(false, a, false, b, (short)0, c,
                                                 false, false);
}

// ---------------------------------------------------------------------------
// Elementwise kernels
// ---------------------------------------------------------------------------
__global__ void add_pe_kernel(const float* __restrict__ q,
                              const float* __restrict__ qa,
                              const float* __restrict__ pe,
                              float* __restrict__ xF, bf16_t* __restrict__ xB,
                              bf16_t* __restrict__ yB) {
  const size_t i = (size_t)blockIdx.x * blockDim.x + threadIdx.x;
  const size_t n = (size_t)B_ * S_ * D_;
  if (i >= n) return;
  const size_t pi = i % ((size_t)S_ * D_);
  const float xv = q[i] + pe[pi];
  const float yv = qa[i] + pe[pi];
  xF[i] = xv;
  xB[i] = f2bf(xv);
  yB[i] = f2bf(yv);
}

__global__ void cvt_bf16_kernel(const float* __restrict__ in,
                                bf16_t* __restrict__ out, size_t n) {
  const size_t i = (size_t)blockIdx.x * blockDim.x + threadIdx.x;
  if (i < n) out[i] = f2bf(in[i]);
}

// v (B,S,D) bf16 -> vT (B,H,DK,S) bf16 so attention B-fragments read contiguous S
__global__ void vtrans_kernel(const bf16_t* __restrict__ v,
                              bf16_t* __restrict__ vt) {
  const size_t i = (size_t)blockIdx.x * blockDim.x + threadIdx.x;
  const size_t n = (size_t)B_ * S_ * D_;
  if (i >= n) return;
  const int b = (int)(i / ((size_t)S_ * D_));
  const int r = (int)(i % ((size_t)S_ * D_));
  const int s = r / D_;
  const int d = r % D_;
  const int h = d / DK_;
  const int dk = d % DK_;
  vt[((size_t)(b * H_ + h) * DK_ + dk) * S_ + s] = v[i];
}

// ---------------------------------------------------------------------------
// bf16 WMMA GEMM: C[M,N] = A[M,K] * W[N,K]^T + bias, optional relu,
// fp32 and/or bf16 outputs. M%128==0, N%64==0, K%32==0.
// 128x64 tile per 256-thread block; each of the 8 waves owns a 2x2 grid of
// 16x16 WMMA subtiles (4 wmma per K-step, A/B fragments reused twice each).
// Global->LDS staging via CDNA5 async-LDS DMA when available, BK=32,
// double-buffered.
// ---------------------------------------------------------------------------
__global__ __launch_bounds__(256) void gemm_bf16_kernel(
    const bf16_t* __restrict__ A, const bf16_t* __restrict__ W,
    const float* __restrict__ bias, float* __restrict__ outF,
    bf16_t* __restrict__ outB, int M, int N, int K, int relu) {
  constexpr int LD = 40;  // 32 + 8 pad (80B rows: 16B aligned, conflict-spread)
  __shared__ bf16_t As[2][128 * LD];
  __shared__ bf16_t Ws[2][64 * LD];

  const int tid  = threadIdx.x;
  const int lane = tid & 31;
  const int wave = tid >> 5;
  const int mBase = blockIdx.x * 128;
  const int nBase = blockIdx.y * 64;

  const int mp = wave & 3;   // -> m subtiles {2*mp, 2*mp+1}
  const int np = wave >> 2;  // -> n subtiles {2*np, 2*np+1}

  // Staging assignment: A tile 128x32 (thread: 16 elems), W tile 64x32 (8 elems)
  const int rowA = tid >> 1;
  const int colA = (tid & 1) * 16;
  const int rowW = tid >> 2;
  const int colW = (tid & 3) * 8;

  const bf16_t* gA = A + (size_t)(mBase + rowA) * K + colA;
  const bf16_t* gW = W + (size_t)(nBase + rowW) * K + colW;

  v8f acc[2][2];
  {
    v8f z = {};
    acc[0][0] = z; acc[0][1] = z; acc[1][0] = z; acc[1][1] = z;
  }

  const int KT = K >> 5;

  // stage tile kt into buffer buf
  auto stage = [&](int buf, int kt) {
    const bf16_t* a = gA + (size_t)kt * 32;
    const bf16_t* w = gW + (size_t)kt * 32;
    copy16_to_lds(a,     &As[buf][rowA * LD + colA]);
    copy16_to_lds(a + 8, &As[buf][rowA * LD + colA + 8]);
    copy16_to_lds(w,     &Ws[buf][rowW * LD + colW]);
  };

  stage(0, 0);
  wait_async_lds();
  __syncthreads();

  const int laneOff = (lane & 15) * LD + (lane >> 4) * 16;  // frag pattern

  for (int kt = 0; kt < KT; ++kt) {
    const int cur = kt & 1;
    if (kt + 1 < KT) stage(cur ^ 1, kt + 1);  // async fill of next buffer

    v16bf a0 = frag_load(&As[cur][(mp * 2 + 0) * 16 * LD + laneOff]);
    v16bf a1 = frag_load(&As[cur][(mp * 2 + 1) * 16 * LD + laneOff]);
    v16bf b0 = frag_load(&Ws[cur][(np * 2 + 0) * 16 * LD + laneOff]);
    v16bf b1 = frag_load(&Ws[cur][(np * 2 + 1) * 16 * LD + laneOff]);
    acc[0][0] = wmma_bf16(a0, b0, acc[0][0]);
    acc[0][1] = wmma_bf16(a0, b1, acc[0][1]);
    acc[1][0] = wmma_bf16(a1, b0, acc[1][0]);
    acc[1][1] = wmma_bf16(a1, b1, acc[1][1]);

    if (kt + 1 < KT) wait_async_lds();
    __syncthreads();
  }

  const int rowHalf = (lane >> 4) * 8;  // C layout: M = j + 8*(lane/16)
  const int colIn   = lane & 15;        // N = lane%16
#pragma unroll
  for (int i = 0; i < 2; ++i) {
#pragma unroll
    for (int u = 0; u < 2; ++u) {
      const int col = nBase + (np * 2 + u) * 16 + colIn;
      const float bv = bias ? bias[col] : 0.f;
#pragma unroll
      for (int j = 0; j < 8; ++j) {
        const int row = mBase + (mp * 2 + i) * 16 + j + rowHalf;
        float v = acc[i][u][j] + bv;
        if (relu) v = fmaxf(v, 0.f);
        const size_t idx = (size_t)row * N + col;
        if (outF) outF[idx] = v;
        if (outB) outB[idx] = f2bf(v);
      }
    }
  }
}

// ---------------------------------------------------------------------------
// Flash-style causal attention (strict lower-triangular, q==k projection).
// Block: (qTile of 64, h, b); 4 waves x 16 query rows.
// Kb: (B,S,D) bf16 K(=Q) projection; Vt: (B,H,DK,S) bf16; Ob: (B,S,D) bf16.
// ---------------------------------------------------------------------------
__global__ __launch_bounds__(128) void attn_kernel(
    const bf16_t* __restrict__ Kb, const bf16_t* __restrict__ Vt,
    bf16_t* __restrict__ Ob) {
  __shared__ bf16_t Pbuf[4][16 * 32];  // per-wave P relayout buffer
  const int lane = threadIdx.x & 31;
  const int wave = threadIdx.x >> 5;
  const int b  = blockIdx.z;
  const int h  = blockIdx.y;
  const int q0 = blockIdx.x * 64 + wave * 16;

  const int rIn     = lane & 15;  // A-frag row / B,C-frag col
  const int seg     = lane >> 4;  // K-half selector
  const int rowHalf = seg * 8;    // C layout row offset

  // Q fragments: 4 chunks of the DK=128 contraction
  v16bf qf[4];
  {
    const bf16_t* qb =
        Kb + (size_t)(b * S_ + q0 + rIn) * D_ + h * DK_ + seg * 16;
#pragma unroll
    for (int c = 0; c < 4; ++c) qf[c] = frag_load(qb + c * 32);
  }

  float m[8], lsum[8];
  v8f acc[8];
  v8f vzero = {};
#pragma unroll
  for (int j = 0; j < 8; ++j) { m[j] = -INFINITY; lsum[j] = 0.f; }
#pragma unroll
  for (int t = 0; t < 8; ++t) acc[t] = vzero;

  const float scale = 0.08838834764831845f;  // 1/sqrt(128)
  bf16_t* Pw = &Pbuf[wave][0];

  for (int k0 = 0; k0 < q0 + 15; k0 += 32) {  // keys needed: k <= q0+14
    v8f s0 = {}, s1 = {};
#pragma unroll
    for (int c = 0; c < 4; ++c) {
      const bf16_t* kb =
          Kb + (size_t)(b * S_ + k0 + rIn) * D_ + h * DK_ + c * 32 + seg * 16;
      v16bf kf0 = frag_load(kb);                     // keys k0..k0+15
      v16bf kf1 = frag_load(kb + (size_t)16 * D_);   // keys k0+16..k0+31
      s0 = wmma_bf16(qf[c], kf0, s0);
      s1 = wmma_bf16(qf[c], kf1, s1);
    }
    float p0[8], p1[8], alpha[8];
#pragma unroll
    for (int j = 0; j < 8; ++j) {
      const int qg = q0 + j + rowHalf;
      float a  = (k0 + rIn      < qg) ? s0[j] * scale : NEGINF;
      float bb = (k0 + 16 + rIn < qg) ? s1[j] * scale : NEGINF;
      float rm = fmaxf(a, bb);
#pragma unroll
      for (int x = 1; x < 16; x <<= 1) rm = fmaxf(rm, __shfl_xor(rm, x, 32));
      const float mn = fmaxf(m[j], rm);
      alpha[j] = __expf(m[j] - mn);
      a  = __expf(a - mn);
      bb = __expf(bb - mn);
      float rs = a + bb;
#pragma unroll
      for (int x = 1; x < 16; x <<= 1) rs += __shfl_xor(rs, x, 32);
      lsum[j] = lsum[j] * alpha[j] + rs;
      m[j] = mn;
      p0[j] = a;
      p1[j] = bb;
    }
#pragma unroll
    for (int t = 0; t < 8; ++t)
#pragma unroll
      for (int j = 0; j < 8; ++j) acc[t][j] *= alpha[j];

    // Re-layout P (C layout -> A fragment) through per-wave LDS
#pragma unroll
    for (int j = 0; j < 8; ++j) {
      Pw[(j + rowHalf) * 32 + rIn]      = f2bf(p0[j]);
      Pw[(j + rowHalf) * 32 + 16 + rIn] = f2bf(p1[j]);
    }
    __builtin_amdgcn_wave_barrier();
    asm volatile("s_wait_dscnt 0" ::: "memory");  // same-wave LDS RAW
    v16bf pf = frag_load(&Pw[rIn * 32 + seg * 16]);

#pragma unroll
    for (int t = 0; t < 8; ++t) {
      const bf16_t* vb = Vt +
          (size_t)((b * H_ + h) * DK_ + t * 16 + rIn) * S_ + k0 + seg * 16;
      v16bf vf = frag_load(vb);
      acc[t] = wmma_bf16(pf, vf, acc[t]);
    }
  }

#pragma unroll
  for (int j = 0; j < 8; ++j) lsum[j] = 1.f / lsum[j];
#pragma unroll
  for (int t = 0; t < 8; ++t) {
#pragma unroll
    for (int j = 0; j < 8; ++j) {
      const int qg = q0 + j + rowHalf;
      float o = acc[t][j] * lsum[j];
      if (qg == 0) o = 0.f;  // reference zero_pad: attn row 0 zeroed
      Ob[(size_t)(b * S_ + qg) * D_ + h * DK_ + t * 16 + rIn] = f2bf(o);
    }
  }
}

// ---------------------------------------------------------------------------
// Fused residual-add + LayerNorm (fp32), emits fp32 residual + bf16 copy
// ---------------------------------------------------------------------------
__global__ __launch_bounds__(256) void resln_kernel(
    const float* __restrict__ res, const float* __restrict__ delta,
    const float* __restrict__ g, const float* __restrict__ bb,
    float* __restrict__ outF, bf16_t* __restrict__ outB) {
  __shared__ float red[256];
  const int row = blockIdx.x;
  const int tid = threadIdx.x;
  const size_t base = (size_t)row * D_;
  float v[4];
  float s = 0.f;
#pragma unroll
  for (int i = 0; i < 4; ++i) {
    const int idx = tid + i * 256;
    v[i] = res[base + idx] + delta[base + idx];
    s += v[i];
  }
  red[tid] = s;
  __syncthreads();
  for (int st = 128; st > 0; st >>= 1) {
    if (tid < st) red[tid] += red[tid + st];
    __syncthreads();
  }
  const float mean = red[0] * (1.f / D_);
  __syncthreads();
  float s2 = 0.f;
#pragma unroll
  for (int i = 0; i < 4; ++i) {
    const float d = v[i] - mean;
    s2 += d * d;
  }
  red[tid] = s2;
  __syncthreads();
  for (int st = 128; st > 0; st >>= 1) {
    if (tid < st) red[tid] += red[tid + st];
    __syncthreads();
  }
  const float rstd = rsqrtf(red[0] * (1.f / D_) + 1e-5f);
#pragma unroll
  for (int i = 0; i < 4; ++i) {
    const int idx = tid + i * 256;
    const float y = (v[i] - mean) * rstd * g[idx] + bb[idx];
    outF[base + idx] = y;
    if (outB) outB[base + idx] = f2bf(y);
  }
}

// ---------------------------------------------------------------------------
extern "C" void kernel_launch(void* const* d_in, const int* in_sizes, int n_in,
                              void* d_out, int out_size, void* d_ws,
                              size_t ws_size, hipStream_t stream) {
  (void)in_sizes; (void)n_in; (void)out_size; (void)ws_size;
  const float* q_embed  = (const float*)d_in[0];
  const float* qa_embed = (const float*)d_in[1];
  const float* pos_emb  = (const float*)d_in[2];
  const float* Wk = (const float*)d_in[3];
  const float* bk = (const float*)d_in[4];
  const float* Wv = (const float*)d_in[5];
  const float* bv = (const float*)d_in[6];
  const float* Wo = (const float*)d_in[7];
  const float* bo = (const float*)d_in[8];
  const float* ln1_g = (const float*)d_in[9];
  const float* ln1_b = (const float*)d_in[10];
  const float* W1 = (const float*)d_in[11];
  const float* b1 = (const float*)d_in[12];
  const float* W2 = (const float*)d_in[13];
  const float* b2 = (const float*)d_in[14];
  const float* ln2_g = (const float*)d_in[15];
  const float* ln2_b = (const float*)d_in[16];

  char* ws = (char*)d_ws;
  size_t off = 0;
  auto alloc = [&](size_t bytes) -> char* {
    char* p = ws + off;
    off += (bytes + 255) & ~(size_t)255;
    return p;
  };
  const size_t nAct = (size_t)M_ * D_;           // 8M elems
  const size_t nHid = (size_t)M_ * FF_;          // 32M elems
  float*  xF   = (float*)alloc(nAct * 4);        // fp32 residual stream
  float*  tmpF = (float*)alloc(nAct * 4);        // pre-LN delta
  bf16_t* xB   = (bf16_t*)alloc(nAct * 2);
  bf16_t* yB   = (bf16_t*)alloc(nAct * 2);
  bf16_t* kB   = (bf16_t*)alloc(nAct * 2);       // K(=Q) projection
  bf16_t* vB   = (bf16_t*)alloc(nAct * 2);       // V projection
  bf16_t* vT   = (bf16_t*)alloc(nAct * 2);       // V transposed (B,H,DK,S)
  bf16_t* aB   = (bf16_t*)alloc(nAct * 2);       // attention output
  bf16_t* hB   = (bf16_t*)alloc(nHid * 2);       // FFN hidden
  bf16_t* WkB  = (bf16_t*)alloc((size_t)L_ * D_ * D_ * 2);
  bf16_t* WvB  = (bf16_t*)alloc((size_t)L_ * D_ * D_ * 2);
  bf16_t* WoB  = (bf16_t*)alloc((size_t)L_ * D_ * D_ * 2);
  bf16_t* W1B  = (bf16_t*)alloc((size_t)L_ * FF_ * D_ * 2);
  bf16_t* W2B  = (bf16_t*)alloc((size_t)L_ * D_ * FF_ * 2);

  const dim3 blk256(256);

  // Embedding + positional encoding (fp32 residual, bf16 activations)
  add_pe_kernel<<<dim3((unsigned)(nAct / 256)), blk256, 0, stream>>>(
      q_embed, qa_embed, pos_emb, xF, xB, yB);

  // One-shot weight conversion to bf16
  const size_t nW  = (size_t)L_ * D_ * D_;
  const size_t nW1 = (size_t)L_ * FF_ * D_;
  cvt_bf16_kernel<<<dim3((unsigned)((nW + 255) / 256)), blk256, 0, stream>>>(Wk, WkB, nW);
  cvt_bf16_kernel<<<dim3((unsigned)((nW + 255) / 256)), blk256, 0, stream>>>(Wv, WvB, nW);
  cvt_bf16_kernel<<<dim3((unsigned)((nW + 255) / 256)), blk256, 0, stream>>>(Wo, WoB, nW);
  cvt_bf16_kernel<<<dim3((unsigned)((nW1 + 255) / 256)), blk256, 0, stream>>>(W1, W1B, nW1);
  cvt_bf16_kernel<<<dim3((unsigned)((nW1 + 255) / 256)), blk256, 0, stream>>>(W2, W2B, nW1);

  const dim3 gD(M_ / 128, D_ / 64);   // 64 x 16
  const dim3 gF(M_ / 128, FF_ / 64);  // 64 x 64
  const dim3 gAttn(S_ / 64, H_, B_);  // 16 x 8 x 8

  for (int l = 0; l < L_; ++l) {
    const size_t wOff  = (size_t)l * D_ * D_;
    const size_t w1Off = (size_t)l * FF_ * D_;

    // K(=Q) and V projections -> bf16
    gemm_bf16_kernel<<<gD, blk256, 0, stream>>>(
        xB, WkB + wOff, bk + (size_t)l * D_, nullptr, kB, M_, D_, D_, 0);
    gemm_bf16_kernel<<<gD, blk256, 0, stream>>>(
        yB, WvB + wOff, bv + (size_t)l * D_, nullptr, vB, M_, D_, D_, 0);
    vtrans_kernel<<<dim3((unsigned)(nAct / 256)), blk256, 0, stream>>>(vB, vT);

    // Causal flash attention
    attn_kernel<<<gAttn, dim3(128), 0, stream>>>(kB, vT, aB);

    // Output projection (fp32 delta), then residual + LN1
    gemm_bf16_kernel<<<gD, blk256, 0, stream>>>(
        aB, WoB + wOff, bo + (size_t)l * D_, tmpF, nullptr, M_, D_, D_, 0);
    resln_kernel<<<dim3(M_), blk256, 0, stream>>>(
        xF, tmpF, ln1_g + (size_t)l * D_, ln1_b + (size_t)l * D_, xF, xB);

    // FFN: D -> FF (ReLU, bf16) -> D (fp32 delta), residual + LN2
    gemm_bf16_kernel<<<gF, blk256, 0, stream>>>(
        xB, W1B + w1Off, b1 + (size_t)l * FF_, nullptr, hB, M_, FF_, D_, 1);
    gemm_bf16_kernel<<<gD, blk256, 0, stream>>>(
        hB, W2B + w1Off, b2 + (size_t)l * D_, tmpF, nullptr, M_, D_, FF_, 0);

    float* lnOut = (l == L_ - 1) ? (float*)d_out : xF;
    resln_kernel<<<dim3(M_), blk256, 0, stream>>>(
        xF, tmpF, ln2_g + (size_t)l * D_, ln2_b + (size_t)l * D_, lnOut, xB);
  }
}
